// MSAD_80985903333740
// MI455X (gfx1250) — matrix operand order repaired and checked by reading, not compiled, not run on hardware
//
#include <hip/hip_runtime.h>
#include <hip/hip_bf16.h>
#include <math.h>

// ---------------------------------------------------------------------------
// CDNA5 (gfx1250) implementation. wave32. WMMA f32_16x16x32_f16 everywhere a
// GEMM appears. Block = 256 threads = 8 waves = (2 row-groups x 4 col-groups),
// block tile = 32 rows x 256 cols, K-step = 32.  Staging uses b128 global
// loads (f32) -> f16 LDS tiles; global_prefetch on the streamed x tiles.
// ---------------------------------------------------------------------------

typedef _Float16 v4h  __attribute__((ext_vector_type(4)));
typedef _Float16 v8h  __attribute__((ext_vector_type(8)));
typedef _Float16 v16h __attribute__((ext_vector_type(16)));
typedef float    v4f  __attribute__((ext_vector_type(4)));
typedef float    v8f  __attribute__((ext_vector_type(8)));

#define EDIM 256
#define LDSK 40      // halfs per row for K=32 staging tiles (+8 pad)
#define LDSE 264     // halfs per row for 256-wide f16 tiles (+8 pad)
#define BM   32      // rows per block tile

__device__ __forceinline__ v8f wmma_f16(v16h a, v16h b, v8f c) {
  return __builtin_amdgcn_wmma_f32_16x16x32_f16(false, a, false, b, (short)0, c,
                                                false, false);
}

// A fragment (16x32, f16).  lane<16: K 0..7 / 16..23 ; lane>=16: K 8..15 / 24..31
__device__ __forceinline__ v16h fragA_lds(const _Float16* s, int stride, int m0,
                                          int k0, int lane) {
  const _Float16* p = s + (size_t)(m0 + (lane & 15)) * stride + k0 + ((lane >> 4) << 3);
  v8h lo = *(const v8h*)p;
  v8h hi = *(const v8h*)(p + 16);
  v16h r;
#pragma unroll
  for (int j = 0; j < 8; ++j) { r[j] = lo[j]; r[8 + j] = hi[j]; }
  return r;
}

// B fragment (32x16, f16), staged as [outCol][k]:  lane<16: K 0..7 / 16..23 ; lane>=16: K 8..15 / 24..31
__device__ __forceinline__ v16h fragB_lds(const _Float16* s, int stride, int n0,
                                          int lane) {
  const _Float16* p = s + (size_t)(n0 + (lane & 15)) * stride + ((lane >> 4) << 3);
  v8h lo = *(const v8h*)p;
  v8h hi = *(const v8h*)(p + 16);
  v16h r;
#pragma unroll
  for (int j = 0; j < 8; ++j) { r[j] = lo[j]; r[8 + j] = hi[j]; }
  return r;
}

// Stage a [rows x 32] f32 tile -> f16 LDS with b128 loads (zero padded OOB).
// All source strides are multiples of 4 floats -> 16B aligned vector loads.
__device__ __forceinline__ void stage_tile(_Float16* dst, const float* src,
                                           int srcStride, int row0, int k0,
                                           int rowMax, int kMax, int rows) {
  for (int t = threadIdx.x; t < rows * 8; t += 256) {
    int r = t >> 3, k4 = (t & 7) << 2;
    int gr = row0 + r, gk = k0 + k4;
    v4f v = {0.f, 0.f, 0.f, 0.f};
    if (gr < rowMax) {
      if (gk + 3 < kMax) {
        v = *(const v4f*)(src + (size_t)gr * srcStride + gk);
      } else {
#pragma unroll
        for (int j = 0; j < 4; ++j)
          if (gk + j < kMax) v[j] = src[(size_t)gr * srcStride + gk + j];
      }
    }
    v4h h;
#pragma unroll
    for (int j = 0; j < 4; ++j) h[j] = (_Float16)v[j];
    *(v4h*)(dst + r * LDSK + k4) = h;
  }
}

// Gated attention score on a 32-row f16 tile sitting in sH:
//   score = (tanh(X aW^T + ab) * sigmoid(X bW^T + bb)) . cW + cb
__device__ __forceinline__ void gated_score_32(
    const _Float16* sH, _Float16* sW, _Float16* sA2, float* sRow,
    const float* aW, const float* ab, const float* bW, const float* bb,
    const float* cW, const float* cb, float* outScore, int row0, int nRows) {
  const int lane = threadIdx.x & 31, w = threadIdx.x >> 5;
  const int m0 = (w >> 2) * 16, nB = (w & 3) * 64;
  v8f acc[4];
#pragma unroll
  for (int t = 0; t < 4; ++t)
#pragma unroll
    for (int j = 0; j < 8; ++j) acc[t][j] = 0.f;

  // pass 1: a = tanh(X aW^T + ab)
  for (int kk = 0; kk < EDIM; kk += 32) {
    stage_tile(sW, aW, EDIM, 0, kk, EDIM, EDIM, 256);
    __syncthreads();
    v16h af = fragA_lds(sH, LDSE, m0, kk, lane);
#pragma unroll
    for (int t = 0; t < 4; ++t) {
      v16h bf = fragB_lds(sW, LDSK, nB + t * 16, lane);
      acc[t] = wmma_f16(af, bf, acc[t]);
    }
    __syncthreads();
  }
#pragma unroll
  for (int t = 0; t < 4; ++t) {
    int n = nB + t * 16 + (lane & 15);
#pragma unroll
    for (int r = 0; r < 8; ++r) {
      int m = m0 + r + ((lane >> 4) << 3);
      sA2[m * LDSE + n] = (_Float16)tanhf(acc[t][r] + ab[n]);
    }
  }
#pragma unroll
  for (int t = 0; t < 4; ++t)
#pragma unroll
    for (int j = 0; j < 8; ++j) acc[t][j] = 0.f;
  __syncthreads();

  // pass 2: gat = sigmoid(X bW^T + bb); per-row reduce of a*gat*cW
  for (int kk = 0; kk < EDIM; kk += 32) {
    stage_tile(sW, bW, EDIM, 0, kk, EDIM, EDIM, 256);
    __syncthreads();
    v16h af = fragA_lds(sH, LDSE, m0, kk, lane);
#pragma unroll
    for (int t = 0; t < 4; ++t) {
      v16h bf = fragB_lds(sW, LDSK, nB + t * 16, lane);
      acc[t] = wmma_f16(af, bf, acc[t]);
    }
    __syncthreads();
  }
  if (threadIdx.x < BM) sRow[threadIdx.x] = 0.f;
  __syncthreads();
#pragma unroll
  for (int t = 0; t < 4; ++t) {
    int n = nB + t * 16 + (lane & 15);
#pragma unroll
    for (int r = 0; r < 8; ++r) {
      int m = m0 + r + ((lane >> 4) << 3);
      float gv = 1.f / (1.f + __expf(-(acc[t][r] + bb[n])));
      float av = (float)sA2[m * LDSE + n];
      atomicAdd(&sRow[m], av * gv * cW[n]);
    }
  }
  __syncthreads();
  if (threadIdx.x < BM && row0 + (int)threadIdx.x < nRows)
    outScore[row0 + threadIdx.x] = sRow[threadIdx.x] + cb[0];
}

// ---------------------------------------------------------------------------
// Kernel 1: genomics path.  gene_snn = gene + bias (written to d_out), then
// g = leaky_relu(ft1(gproj(gene_snn))) -> ws.
// ---------------------------------------------------------------------------
__global__ void __launch_bounds__(256)
gene_kernel(const float* __restrict__ gene, const float* __restrict__ gbias,
            const float* __restrict__ gpW, const float* __restrict__ gpb,
            const float* __restrict__ ftW, const float* __restrict__ ftb,
            float* __restrict__ outSnn, float* __restrict__ gOut) {
  __shared__ __align__(16) _Float16 sX[BM * LDSK];
  __shared__ __align__(16) _Float16 sW[256 * LDSK];
  __shared__ __align__(16) _Float16 sH[BM * LDSE];
  const int lane = threadIdx.x & 31, w = threadIdx.x >> 5;
  const int m0 = (w >> 2) * 16, nB = (w & 3) * 64;
  const int row0 = blockIdx.x * BM;
  v8f acc[4];
#pragma unroll
  for (int t = 0; t < 4; ++t)
#pragma unroll
    for (int j = 0; j < 8; ++j) acc[t][j] = 0.f;

  for (int kk = 0; kk < 200; kk += 32) {
    // stage gene_snn tile (with d_out writeback), b128 loads where possible
    for (int t = threadIdx.x; t < BM * 8; t += 256) {
      int r = t >> 3, k4 = (t & 7) << 2;
      int gr = row0 + r, gk = kk + k4;
      v4f v = {0.f, 0.f, 0.f, 0.f};
      if (gr < 10000) {
        if (gk + 3 < 200) {
          v4f a = *(const v4f*)(gene + (size_t)gr * 200 + gk);
          v4f b = *(const v4f*)(gbias + (size_t)gr * 200 + gk);
#pragma unroll
          for (int j = 0; j < 4; ++j) v[j] = a[j] + b[j];
          *(v4f*)(outSnn + (size_t)gr * 200 + gk) = v;
        } else {
#pragma unroll
          for (int j = 0; j < 4; ++j)
            if (gk + j < 200) {
              v[j] = gene[(size_t)gr * 200 + gk + j] +
                     gbias[(size_t)gr * 200 + gk + j];
              outSnn[(size_t)gr * 200 + gk + j] = v[j];
            }
        }
      }
      v4h hh;
#pragma unroll
      for (int j = 0; j < 4; ++j) hh[j] = (_Float16)v[j];
      *(v4h*)(sX + r * LDSK + k4) = hh;
    }
    stage_tile(sW, gpW, 200, 0, kk, EDIM, 200, 256);
    __syncthreads();
    v16h af = fragA_lds(sX, LDSK, m0, 0, lane);
#pragma unroll
    for (int t = 0; t < 4; ++t) {
      v16h bf = fragB_lds(sW, LDSK, nB + t * 16, lane);
      acc[t] = wmma_f16(af, bf, acc[t]);
    }
    __syncthreads();
  }
#pragma unroll
  for (int t = 0; t < 4; ++t) {
    int n = nB + t * 16 + (lane & 15);
#pragma unroll
    for (int r = 0; r < 8; ++r) {
      int m = m0 + r + ((lane >> 4) << 3);
      sH[m * LDSE + n] = (_Float16)(acc[t][r] + gpb[n]);
    }
  }
#pragma unroll
  for (int t = 0; t < 4; ++t)
#pragma unroll
    for (int j = 0; j < 8; ++j) acc[t][j] = 0.f;
  __syncthreads();

  for (int kk = 0; kk < EDIM; kk += 32) {
    stage_tile(sW, ftW, EDIM, 0, kk, EDIM, EDIM, 256);
    __syncthreads();
    v16h af = fragA_lds(sH, LDSE, m0, kk, lane);
#pragma unroll
    for (int t = 0; t < 4; ++t) {
      v16h bf = fragB_lds(sW, LDSK, nB + t * 16, lane);
      acc[t] = wmma_f16(af, bf, acc[t]);
    }
    __syncthreads();
  }
#pragma unroll
  for (int t = 0; t < 4; ++t) {
    int n = nB + t * 16 + (lane & 15);
#pragma unroll
    for (int r = 0; r < 8; ++r) {
      int m = m0 + r + ((lane >> 4) << 3);
      int row = row0 + m;
      if (row < 10000) {
        float v = acc[t][r] + ftb[n];
        gOut[(size_t)row * EDIM + n] = (v > 0.f) ? v : 0.01f * v;
      }
    }
  }
}

// ---------------------------------------------------------------------------
// Kernel 2: fused WSI projection + gated attention score (never writes h).
// ---------------------------------------------------------------------------
__global__ void __launch_bounds__(256)
wsi_score_kernel(const float* __restrict__ x, int N,
                 const float* __restrict__ wsiW, const float* __restrict__ wsib,
                 const float* aW, const float* ab, const float* bW,
                 const float* bb, const float* cW, const float* cb,
                 float* __restrict__ outScore) {
  __shared__ __align__(16) _Float16 sX[BM * LDSK];
  __shared__ __align__(16) _Float16 sW[256 * LDSK];
  __shared__ __align__(16) _Float16 sH[BM * LDSE];
  __shared__ __align__(16) _Float16 sA2[BM * LDSE];
  __shared__ float sRow[BM];
  const int lane = threadIdx.x & 31, w = threadIdx.x >> 5;
  const int m0 = (w >> 2) * 16, nB = (w & 3) * 64;
  const int row0 = blockIdx.x * BM;
  v8f acc[4];
#pragma unroll
  for (int t = 0; t < 4; ++t)
#pragma unroll
    for (int j = 0; j < 8; ++j) acc[t][j] = 0.f;

  for (int kk = 0; kk < 768; kk += 32) {
    // prefetch next K-tile of the HBM-streamed x rows (global_prefetch_b8)
    if (kk + 32 < 768)
      __builtin_prefetch(x + (size_t)(row0 + (threadIdx.x & 31)) * 768 + kk + 32,
                         0, 1);
    stage_tile(sX, x, 768, row0, kk, N, 768, BM);
    stage_tile(sW, wsiW, 768, 0, kk, EDIM, 768, 256);
    __syncthreads();
    v16h af = fragA_lds(sX, LDSK, m0, 0, lane);
#pragma unroll
    for (int t = 0; t < 4; ++t) {
      v16h bf = fragB_lds(sW, LDSK, nB + t * 16, lane);
      acc[t] = wmma_f16(af, bf, acc[t]);
    }
    __syncthreads();
  }
#pragma unroll
  for (int t = 0; t < 4; ++t) {
    int n = nB + t * 16 + (lane & 15);
#pragma unroll
    for (int r = 0; r < 8; ++r) {
      int m = m0 + r + ((lane >> 4) << 3);
      float h = acc[t][r] + wsib[n];
      sH[m * LDSE + n] = (_Float16)(h > 0.f ? h : 0.f);
    }
  }
  __syncthreads();
  gated_score_32(sH, sW, sA2, sRow, aW, ab, bW, bb, cW, cb, outScore, row0, N);
}

// ---------------------------------------------------------------------------
// Kernel 3: top-6 per path (argmax x6 with lowest-index tie break, like top_k).
// ---------------------------------------------------------------------------
__global__ void __launch_bounds__(256)
topk_kernel(const float* __restrict__ sc, int n0, int n1, int n2,
            int* __restrict__ idxOut) {
  __shared__ float bv[256];
  __shared__ int bi[256];
  __shared__ int chosen[6];
  int p = blockIdx.x;
  int off = (p == 0) ? 0 : ((p == 1) ? n0 : (n0 + n1));
  int N = (p == 0) ? n0 : ((p == 1) ? n1 : n2);
  const float* s = sc + off;
  for (int kp = 0; kp < 6; ++kp) {
    float best = -INFINITY;
    int besti = 0x7fffffff;
    for (int n = threadIdx.x; n < N; n += 256) {
      bool skip = false;
      for (int j = 0; j < kp; ++j)
        if (chosen[j] == n) skip = true;
      if (!skip) {
        float v = s[n];
        if (v > best || (v == best && n < besti)) { best = v; besti = n; }
      }
    }
    bv[threadIdx.x] = best; bi[threadIdx.x] = besti;
    __syncthreads();
    for (int st = 128; st > 0; st >>= 1) {
      if ((int)threadIdx.x < st) {
        float ov = bv[threadIdx.x + st]; int oi = bi[threadIdx.x + st];
        if (ov > bv[threadIdx.x] ||
            (ov == bv[threadIdx.x] && oi < bi[threadIdx.x])) {
          bv[threadIdx.x] = ov; bi[threadIdx.x] = oi;
        }
      }
      __syncthreads();
    }
    if (threadIdx.x == 0) { chosen[kp] = bi[0]; idxOut[p * 6 + kp] = bi[0]; }
    __syncthreads();
  }
}

// ---------------------------------------------------------------------------
// Kernel 4: recompute h for selected rows, then k/v = xs @ {Wk,Wv}^T + b.
// ---------------------------------------------------------------------------
__global__ void __launch_bounds__(256)
gather_kv_kernel(const float* x5, const float* x10, const float* x20,
                 const int* __restrict__ idx, const float* wsiW,
                 const float* wsib, const float* coInW, const float* coInB,
                 float* xs, float* kmat, float* vmat) {
  int i = blockIdx.x;
  int e = threadIdx.x;
  const float* x = (i == 0) ? x5 : ((i == 1) ? x10 : x20);
  __shared__ float xsh[6][EDIM];
  for (int j = 0; j < 6; ++j) {
    int row = idx[i * 6 + j];
    const float* xr = x + (size_t)row * 768;
    const float* wr = wsiW + (size_t)e * 768;
    float a = wsib[e];
    for (int c = 0; c < 768; ++c) a += xr[c] * wr[c];
    a = a > 0.f ? a : 0.f;
    xsh[j][e] = a;
    xs[(size_t)(i * 6 + j) * EDIM + e] = a;
  }
  __syncthreads();
  for (int j = 0; j < 6; ++j) {
    float ak = coInB[(size_t)i * 768 + 256 + e];
    float av = coInB[(size_t)i * 768 + 512 + e];
    const float* wk = coInW + ((size_t)i * 768 + 256 + e) * EDIM;
    const float* wv = coInW + ((size_t)i * 768 + 512 + e) * EDIM;
    for (int c = 0; c < EDIM; ++c) {
      float xv = xsh[j][c];
      ak += xv * wk[c];
      av += xv * wv[c];
    }
    kmat[(size_t)(i * 6 + j) * EDIM + e] = ak;
    vmat[(size_t)(i * 6 + j) * EDIM + e] = av;
  }
}

// ---------------------------------------------------------------------------
// Kernel 5: co-attention.  q=g@Wq^T*scale ; attn=softmax(q k^T) ; o=attn v ;
// co = o@Wout^T + b.
// ---------------------------------------------------------------------------
__global__ void __launch_bounds__(256)
coattn_kernel(const float* __restrict__ g, const float* __restrict__ coInW,
              const float* __restrict__ coInB, const float* __restrict__ coOutW,
              const float* __restrict__ coOutB, const float* __restrict__ kmat,
              const float* __restrict__ vmat, float* __restrict__ co) {
  __shared__ __align__(16) _Float16 sX[BM * LDSK];
  __shared__ __align__(16) _Float16 sW[256 * LDSK];
  __shared__ __align__(16) _Float16 sH[BM * LDSE];
  __shared__ __align__(16) _Float16 sA2[BM * LDSE];
  __shared__ float sAttn[BM][8];
  const int i = blockIdx.y;
  const int row0 = blockIdx.x * BM;
  const int lane = threadIdx.x & 31, w = threadIdx.x >> 5;
  const int m0 = (w >> 2) * 16, nB = (w & 3) * 64;
  const float* Wq = coInW + (size_t)i * 768 * EDIM;
  const float* bq = coInB + (size_t)i * 768;
  const float scale = 0.0625f;  // 256^-0.5
  v8f acc[4];
#pragma unroll
  for (int t = 0; t < 4; ++t)
#pragma unroll
    for (int j = 0; j < 8; ++j) acc[t][j] = 0.f;

  for (int kk = 0; kk < EDIM; kk += 32) {
    stage_tile(sX, g, EDIM, row0, kk, 10000, EDIM, BM);
    stage_tile(sW, Wq, EDIM, 0, kk, EDIM, EDIM, 256);
    __syncthreads();
    v16h af = fragA_lds(sX, LDSK, m0, 0, lane);
#pragma unroll
    for (int t = 0; t < 4; ++t) {
      v16h bf = fragB_lds(sW, LDSK, nB + t * 16, lane);
      acc[t] = wmma_f16(af, bf, acc[t]);
    }
    __syncthreads();
  }
#pragma unroll
  for (int t = 0; t < 4; ++t) {
    int n = nB + t * 16 + (lane & 15);
#pragma unroll
    for (int r = 0; r < 8; ++r) {
      int m = m0 + r + ((lane >> 4) << 3);
      sH[m * LDSE + n] = (_Float16)((acc[t][r] + bq[n]) * scale);
    }
  }
  __syncthreads();
  if ((int)threadIdx.x < BM * 6) {
    int r = threadIdx.x / 6, j = threadIdx.x % 6;
    const float* kr = kmat + ((size_t)i * 6 + j) * EDIM;
    float a = 0.f;
    for (int c = 0; c < EDIM; ++c) a += (float)sH[r * LDSE + c] * kr[c];
    sAttn[r][j] = a;
  }
  __syncthreads();
  if (threadIdx.x < BM) {
    float mx = sAttn[threadIdx.x][0];
    for (int j = 1; j < 6; ++j) mx = fmaxf(mx, sAttn[threadIdx.x][j]);
    float s = 0.f;
    for (int j = 0; j < 6; ++j) {
      float e2 = __expf(sAttn[threadIdx.x][j] - mx);
      sAttn[threadIdx.x][j] = e2;
      s += e2;
    }
    float inv = 1.f / s;
    for (int j = 0; j < 6; ++j) sAttn[threadIdx.x][j] *= inv;
  }
  __syncthreads();
  for (int t = threadIdx.x; t < BM * EDIM; t += 256) {
    int r = t >> 8, e2 = t & 255;
    float o = 0.f;
    for (int j = 0; j < 6; ++j)
      o += sAttn[r][j] * vmat[((size_t)i * 6 + j) * EDIM + e2];
    sA2[r * LDSE + e2] = (_Float16)o;
  }
#pragma unroll
  for (int t = 0; t < 4; ++t)
#pragma unroll
    for (int j = 0; j < 8; ++j) acc[t][j] = 0.f;
  __syncthreads();
  const float* Wo = coOutW + (size_t)i * EDIM * EDIM;
  const float* bo = coOutB + (size_t)i * EDIM;
  for (int kk = 0; kk < EDIM; kk += 32) {
    stage_tile(sW, Wo, EDIM, 0, kk, EDIM, EDIM, 256);
    __syncthreads();
    v16h af = fragA_lds(sA2, LDSE, m0, kk, lane);
#pragma unroll
    for (int t = 0; t < 4; ++t) {
      v16h bf = fragB_lds(sW, LDSK, nB + t * 16, lane);
      acc[t] = wmma_f16(af, bf, acc[t]);
    }
    __syncthreads();
  }
#pragma unroll
  for (int t = 0; t < 4; ++t) {
    int n = nB + t * 16 + (lane & 15);
#pragma unroll
    for (int r = 0; r < 8; ++r) {
      int m = m0 + r + ((lane >> 4) << 3);
      int row = row0 + m;
      if (row < 10000)
        co[((size_t)i * 10000 + row) * EDIM + n] = acc[t][r] + bo[n];
    }
  }
}

// ---------------------------------------------------------------------------
// Kernel 6: gated attention scores on co (3 x 10000 rows).
// ---------------------------------------------------------------------------
__global__ void __launch_bounds__(256)
ga_score_kernel(const float* __restrict__ co, const float* aW, const float* ab,
                const float* bW, const float* bb, const float* cW,
                const float* cb, float* __restrict__ gsc) {
  __shared__ __align__(16) _Float16 sW[256 * LDSK];
  __shared__ __align__(16) _Float16 sH[BM * LDSE];
  __shared__ __align__(16) _Float16 sA2[BM * LDSE];
  __shared__ float sRow[BM];
  const int i = blockIdx.y;
  const int row0 = blockIdx.x * BM;
  const float* X = co + (size_t)i * 10000 * EDIM;
  for (int t = threadIdx.x; t < BM * (EDIM / 4); t += 256) {
    int r = t >> 6, c4 = (t & 63) << 2;
    v4f v = {0.f, 0.f, 0.f, 0.f};
    if (row0 + r < 10000)
      v = *(const v4f*)(X + (size_t)(row0 + r) * EDIM + c4);
    v4h h;
#pragma unroll
    for (int j = 0; j < 4; ++j) h[j] = (_Float16)v[j];
    *(v4h*)(sH + r * LDSE + c4) = h;
  }
  __syncthreads();
  gated_score_32(sH, sW, sA2, sRow, aW, ab, bW, bb, cW, cb,
                 gsc + (size_t)i * 10000, row0, 10000);
}

// ---------------------------------------------------------------------------
// Kernel 7: softmax over 10000 + attention-weighted pooling -> h_path (3x256).
// ---------------------------------------------------------------------------
__global__ void __launch_bounds__(256)
pool_kernel(const float* __restrict__ gsc, const float* __restrict__ co,
            float* __restrict__ hp) {
  int i = blockIdx.x;
  const float* s = gsc + (size_t)i * 10000;
  __shared__ float red[256];
  __shared__ float wts[256];
  __shared__ float stat[2];
  float mx = -INFINITY;
  for (int n = threadIdx.x; n < 10000; n += 256) mx = fmaxf(mx, s[n]);
  red[threadIdx.x] = mx;
  __syncthreads();
  for (int st = 128; st > 0; st >>= 1) {
    if ((int)threadIdx.x < st)
      red[threadIdx.x] = fmaxf(red[threadIdx.x], red[threadIdx.x + st]);
    __syncthreads();
  }
  if (threadIdx.x == 0) stat[0] = red[0];
  __syncthreads();
  float m = stat[0];
  float se = 0.f;
  for (int n = threadIdx.x; n < 10000; n += 256) se += __expf(s[n] - m);
  __syncthreads();
  red[threadIdx.x] = se;
  __syncthreads();
  for (int st = 128; st > 0; st >>= 1) {
    if ((int)threadIdx.x < st) red[threadIdx.x] += red[threadIdx.x + st];
    __syncthreads();
  }
  if (threadIdx.x == 0) stat[1] = 1.f / red[0];
  __syncthreads();
  float inv = stat[1];
  float acc = 0.f;
  for (int base = 0; base < 10000; base += 256) {
    int lim = 10000 - base; if (lim > 256) lim = 256;
    __syncthreads();
    int n = base + threadIdx.x;
    wts[threadIdx.x] = (n < 10000) ? __expf(s[n] - m) * inv : 0.f;
    __syncthreads();
    for (int n2 = 0; n2 < lim; ++n2)
      acc += wts[n2] * co[((size_t)i * 10000 + base + n2) * EDIM + threadIdx.x];
  }
  hp[(size_t)i * EDIM + threadIdx.x] = acc;
}

// ---------------------------------------------------------------------------
// Kernel 8: tiny head: rho, wf gates, classifier, sigmoid, cumprod, argmax.
// ---------------------------------------------------------------------------
__global__ void __launch_bounds__(256)
head_kernel(const float* __restrict__ hp, const float* rhoW, const float* rhob,
            const float* W5, const float* b5, const float* W10,
            const float* b10, const float* clsW, const float* clsb,
            float* __restrict__ out) {
  __shared__ float h[3][EDIM];
  __shared__ float red[256];
  __shared__ float cgate[2];
  __shared__ float h20s[EDIM];
  __shared__ float sc[4];
  int e = threadIdx.x;
  for (int i = 0; i < 3; ++i) {
    const float* wr = rhoW + (size_t)e * EDIM;
    const float* x = hp + (size_t)i * EDIM;
    float a = rhob[e];
    for (int c = 0; c < EDIM; ++c) a += x[c] * wr[c];
    h[i][e] = a > 0.f ? a : 0.f;
  }
  __syncthreads();
  red[e] = h[1][e] * W5[e];
  __syncthreads();
  for (int st = 128; st > 0; st >>= 1) {
    if (e < st) red[e] += red[e + st];
    __syncthreads();
  }
  if (e == 0) cgate[0] = 1.f / (1.f + __expf(-(red[0] + b5[0])));
  __syncthreads();
  float h10n = h[1][e] + cgate[0] * h[0][e];
  red[e] = h[2][e] * W10[e];
  __syncthreads();
  for (int st = 128; st > 0; st >>= 1) {
    if (e < st) red[e] += red[e + st];
    __syncthreads();
  }
  if (e == 0) cgate[1] = 1.f / (1.f + __expf(-(red[0] + b10[0])));
  __syncthreads();
  float h20n = h[2][e] + cgate[1] * h10n;
  out[9 + e] = h20n;
  h20s[e] = h20n;
  __syncthreads();
  if (e < 4) {
    float l = clsb[e];
    for (int c = 0; c < EDIM; ++c) l += h20s[c] * clsW[e * EDIM + c];
    sc[e] = l;
  }
  __syncthreads();
  if (e == 0) {
    int best = 0; float bv = sc[0];
    for (int j = 1; j < 4; ++j)
      if (sc[j] > bv) { bv = sc[j]; best = j; }
    float S = 1.f;
    for (int j = 0; j < 4; ++j) {
      float hz = 1.f / (1.f + __expf(-sc[j]));
      out[j] = hz;
      S *= (1.f - hz);
      out[4 + j] = S;
    }
    out[8] = (float)best;
  }
}

// ---------------------------------------------------------------------------
extern "C" void kernel_launch(void* const* d_in, const int* in_sizes, int n_in,
                              void* d_out, int out_size, void* d_ws,
                              size_t ws_size, hipStream_t stream) {
  (void)in_sizes; (void)n_in; (void)out_size; (void)ws_size;
  const float* x5    = (const float*)d_in[0];
  const float* x10   = (const float*)d_in[1];
  const float* x20   = (const float*)d_in[2];
  const float* gene  = (const float*)d_in[3];
  const float* gbias = (const float*)d_in[4];
  const float* wsiW  = (const float*)d_in[5];
  const float* wsib  = (const float*)d_in[6];
  const float* gpW   = (const float*)d_in[7];
  const float* gpb   = (const float*)d_in[8];
  const float* ftW   = (const float*)d_in[9];
  const float* ftb   = (const float*)d_in[10];
  const float* anaW  = (const float*)d_in[11];
  const float* anab  = (const float*)d_in[12];
  const float* anbW  = (const float*)d_in[13];
  const float* anbb  = (const float*)d_in[14];
  const float* ancW  = (const float*)d_in[15];
  const float* ancb  = (const float*)d_in[16];
  const float* coInW = (const float*)d_in[17];
  const float* coInB = (const float*)d_in[18];
  const float* coOutW= (const float*)d_in[19];
  const float* coOutB= (const float*)d_in[20];
  const float* gaaW  = (const float*)d_in[21];
  const float* gaab  = (const float*)d_in[22];
  const float* gabW  = (const float*)d_in[23];
  const float* gabb  = (const float*)d_in[24];
  const float* gacW  = (const float*)d_in[25];
  const float* gacb  = (const float*)d_in[26];
  const float* rhoW  = (const float*)d_in[27];
  const float* rhob  = (const float*)d_in[28];
  const float* W5    = (const float*)d_in[29];
  const float* b5    = (const float*)d_in[30];
  const float* W10   = (const float*)d_in[31];
  const float* b10   = (const float*)d_in[32];
  const float* clsW  = (const float*)d_in[33];
  const float* clsb  = (const float*)d_in[34];

  float* out = (float*)d_out;
  float* ws = (float*)d_ws;

  const size_t OFF_G   = 0;                                  // 10000*256
  const size_t OFF_CO  = OFF_G + (size_t)10000 * 256;        // 3*10000*256
  const size_t OFF_SC  = OFF_CO + (size_t)3 * 10000 * 256;   // 106496
  const size_t OFF_GSC = OFF_SC + (8192 + 32768 + 65536);    // 30000
  const size_t OFF_IDX = OFF_GSC + 30000;                    // 18 ints (pad 32)
  const size_t OFF_XS  = OFF_IDX + 32;                       // 3*6*256
  const size_t OFF_K   = OFF_XS + (size_t)3 * 6 * 256;
  const size_t OFF_V   = OFF_K + (size_t)3 * 6 * 256;
  const size_t OFF_HP  = OFF_V + (size_t)3 * 6 * 256;

  float* gW  = ws + OFF_G;
  float* coW = ws + OFF_CO;
  float* scW = ws + OFF_SC;
  float* gscW= ws + OFF_GSC;
  int*   idxW= (int*)(ws + OFF_IDX);
  float* xsW = ws + OFF_XS;
  float* kW  = ws + OFF_K;
  float* vW  = ws + OFF_V;
  float* hpW = ws + OFF_HP;

  gene_kernel<<<313, 256, 0, stream>>>(gene, gbias, gpW, gpb, ftW, ftb,
                                       out + 265, gW);
  wsi_score_kernel<<<8192 / BM, 256, 0, stream>>>(
      x5, 8192, wsiW, wsib, anaW, anab, anbW, anbb, ancW, ancb, scW);
  wsi_score_kernel<<<32768 / BM, 256, 0, stream>>>(
      x10, 32768, wsiW, wsib, anaW, anab, anbW, anbb, ancW, ancb, scW + 8192);
  wsi_score_kernel<<<65536 / BM, 256, 0, stream>>>(
      x20, 65536, wsiW, wsib, anaW, anab, anbW, anbb, ancW, ancb,
      scW + 8192 + 32768);
  topk_kernel<<<3, 256, 0, stream>>>(scW, 8192, 32768, 65536, idxW);
  gather_kv_kernel<<<3, 256, 0, stream>>>(x5, x10, x20, idxW, wsiW, wsib,
                                          coInW, coInB, xsW, kW, vW);
  coattn_kernel<<<dim3(313, 3), 256, 0, stream>>>(gW, coInW, coInB, coOutW,
                                                  coOutB, kW, vW, coW);
  ga_score_kernel<<<dim3(313, 3), 256, 0, stream>>>(coW, gaaW, gaab, gabW,
                                                    gabb, gacW, gacb, gscW);
  pool_kernel<<<3, 256, 0, stream>>>(gscW, coW, hpW);
  head_kernel<<<1, 256, 0, stream>>>(hpW, rhoW, rhob, W5, b5, W10, b10, clsW,
                                     clsb, out);
}
